// CLF_QP_Net_2001454760659
// MI455X (gfx1250) — matrix operand order repaired and checked by reading, not compiled, not run on hardware
//
#include <hip/hip_runtime.h>

// ---------------------------------------------------------------------------
// CLF-QP network, fused for MI455X (gfx1250, wave32, WMMA + async-LDS DMA).
//   GEMM path: v_wmma_f32_16x16x32_f16, f32 accumulate.
//   One wave handles a 16-row batch tile; a1/t2 tiles live in LDS.
//   W2 N-panels are block-shared, double-buffered in LDS, filled with
//   global_load_async_to_lds_b128 (ASYNCcnt) to cut L2 traffic 8x.
//   nt loop unrolled x2 so both panel buffers are static pointers and all
//   DS accesses use immediate offsets; tanh uses v_tanh_f32 when available.
// ---------------------------------------------------------------------------

typedef __attribute__((ext_vector_type(16))) _Float16 v16h;
typedef __attribute__((ext_vector_type(8)))  float    v8f;

#define B_TOTAL   65536
#define NHID      512
#define WPB       8                  // waves per block (256 threads)
#define NBLOCKS   ((B_TOTAL / 16) / WPB)            // 512
#define PITCH_H   520                // halfs per LDS row (512 + 8 pad)
#define PITCH_B   (PITCH_H * 2)      // 1040 B; 260 dwords -> bank stride 4
#define TILE_BYTES (16 * PITCH_B)    // 16640 B per 16x512 f16 tile
#define WAVE_LDS  (2 * TILE_BYTES)   // a1 tile + t2 tile
#define BPANEL_B  TILE_BYTES         // one 16x512 f16 B panel (padded)
#define SHARED_HDR ((1024 + 512 + 512 + WPB * 48) * 4)    // 9728 B
#define BBUF_OFF  (SHARED_HDR + WPB * WAVE_LDS)           // 275,968 B
#define LDS_TOTAL (BBUF_OFF + 2 * BPANEL_B)               // 309,248 B

union FragU { v16h v; uint4 q[2]; };
union PairU { _Float16 h[2]; unsigned u; };

__device__ __forceinline__ float fast_tanh(float x) {
#if __has_builtin(__builtin_amdgcn_tanhf)
    return __builtin_amdgcn_tanhf(x);             // v_tanh_f32 (TRANS32)
#else
    // tanh(x) = 1 - 2/(exp(2x)+1); exp via v_exp_f32 (2^x). Saturates cleanly.
    float e = __builtin_amdgcn_exp2f(x * 2.8853900817779268f); // 2*log2(e)
    return 1.0f - 2.0f * __builtin_amdgcn_rcpf(e + 1.0f);
#endif
}

// Cooperatively fill one 16x512-half B panel (padded pitch) in LDS from
// `base` (f16, row-major [512][512]) rows nt*16..nt*16+15, using the CDNA5
// async LDS-DMA path. 1024 16B chunks / 256 threads = 4 per thread.
__device__ __forceinline__ void issue_panel(const _Float16* __restrict__ base,
                                            int nt, int buf,
                                            unsigned ldsBBuf, int tid) {
#pragma unroll
    for (int c = 0; c < 4; ++c) {
        int flat = c * 256 + tid;            // 16B chunk index; 64 per row
        int row  = flat >> 6;
        int col  = flat & 63;
        unsigned loff = ldsBBuf + (unsigned)(buf * BPANEL_B + row * PITCH_B + col * 16);
        unsigned goff = (unsigned)((nt * 16 + row) * 1024 + col * 16);
        asm volatile("global_load_async_to_lds_b128 %0, %1, %2"
                     :: "v"(loff), "v"(goff), "s"(base)
                     : "memory");
    }
}

// Pre-pass: W2 (f32 row-major) -> f16 row-major (B for a1@W2^T)
//                              -> f16 transposed (B for t2@W2)
__global__ void convert_w2(const float* __restrict__ W2,
                           _Float16* __restrict__ w2h,
                           _Float16* __restrict__ w2th) {
    int idx = blockIdx.x * 256 + threadIdx.x;     // 1024 blocks x 256 = 262144
    int r = idx >> 9, c = idx & 511;
    float v = W2[idx];
    w2h[idx]          = (_Float16)v;              // w2h[n][k]  = W2[n][k]
    w2th[c * 512 + r] = (_Float16)v;              // w2th[n][k] = W2[k][n]
}

__global__ __launch_bounds__(256, 1)
void clf_qp_fused(const float* __restrict__ x,  const float* __restrict__ W1,
                  const float* __restrict__ b1, const float* __restrict__ b2,
                  const float* __restrict__ K,
                  const _Float16* __restrict__ w2h,
                  const _Float16* __restrict__ w2th,
                  float* __restrict__ out) {
    extern __shared__ char smem[];
    float* W1s = (float*)smem;          // [512][2]
    float* b1s = W1s + 1024;            // [512]
    float* b2s = b1s + 512;             // [512]
    float* red = b2s + 512;             // [WPB][16][3]
    char*  waveArea = smem + SHARED_HDR;
    char*  bbuf     = smem + BBUF_OFF;  // [2] x 16x512 f16 panel, padded

    const int tid = threadIdx.x;
    const unsigned ldsBBuf = __builtin_amdgcn_groupstaticsize() + BBUF_OFF;

    // Kick off the first GEMM1 B panel before anything else (overlaps stage 1).
    issue_panel(w2h, 0, 0, ldsBBuf, tid);

    for (int i = tid; i < 1024; i += 256) W1s[i] = W1[i];
    for (int i = tid; i < 512;  i += 256) { b1s[i] = b1[i]; b2s[i] = b2[i]; }
    __syncthreads();

    const int wave = tid >> 5;
    const int lane = tid & 31;
    const int l15  = lane & 15;
    const int half = lane >> 4;
    const int tile = blockIdx.x * WPB + wave;

    char* a1b = waveArea + wave * WAVE_LDS;   // 16x512 f16, pitch 1040B
    char* t2b = a1b + TILE_BYTES;

    // ---- Stage 1: a1 = tanh(x @ W1^T + b1), f16 row-major in LDS ----------
    {
        const int m = l15;
        const int g = tile * 16 + m;
        const float x0 = x[2 * g], x1 = x[2 * g + 1];
        for (int i = 0; i < 128; ++i) {
            int p = 2 * i + half;           // halves interleave f16 pairs
            int k = 2 * p;
            float w00 = W1s[2 * k + 0], w01 = W1s[2 * k + 1];
            float w10 = W1s[2 * k + 2], w11 = W1s[2 * k + 3];
            float v0 = fast_tanh(x0 * w00 + x1 * w01 + b1s[k]);
            float v1 = fast_tanh(x0 * w10 + x1 * w11 + b1s[k + 1]);
            PairU pk; pk.h[0] = (_Float16)v0; pk.h[1] = (_Float16)v1;
            *(unsigned*)(a1b + m * PITCH_B + k * 2) = pk.u;
        }
    }
    asm volatile("s_wait_dscnt 0" ::: "memory");   // a1 tile visible wave-wide

    float vacc[8], g0a[8], g1a[8];
#pragma unroll
    for (int d = 0; d < 8; ++d) { vacc[d] = 0.f; g0a[d] = 0.f; g1a[d] = 0.f; }

    // Fragment bases: A reads row l15 (K chunk by half); B reads panel row l15.
    const int abyte0 = l15 * PITCH_B + half * 16;   // K halfs: half*8
    const int bbyte0 = l15 * PITCH_B + half * 16;   // K halfs: half*8
    const char* bp0 = bbuf + bbyte0;                // static panel pointers
    const char* bp1 = bbuf + BPANEL_B + bbyte0;

    // One 16x16x512 WMMA sweep: C += Atile @ panel^T (K immediates off bp).
    auto mma_sweep = [&](const char* at, const char* bp) -> v8f {
        v8f acc = {};
#pragma unroll 4
        for (int kt = 0; kt < 16; ++kt) {
            FragU A, Bf;
            A.q[0]  = *(const uint4*)(at + kt * 64);
            A.q[1]  = *(const uint4*)(at + kt * 64 + 32);
            Bf.q[0] = *(const uint4*)(bp + kt * 64);
            Bf.q[1] = *(const uint4*)(bp + kt * 64 + 32);
            acc = __builtin_amdgcn_wmma_f32_16x16x32_f16(
                      false, A.v, false, Bf.v, (short)0, acc, false, false);
        }
        return acc;
    };

    // GEMM1 epilogue: a2 = tanh(acc + b2); V partial; t2 -> LDS (A layout).
    auto gemm1_epi = [&](v8f acc, int nt) {
        const int n  = nt * 16 + l15;
        const float bb = b2s[n];
#pragma unroll
        for (int d = 0; d < 8; ++d) {
            float a2 = fast_tanh(acc[d] + bb);
            vacc[d] += a2 * a2;
            float t2 = a2 * (1.0f - a2 * a2);
            int m = d + 8 * half;                     // D layout row
            *(_Float16*)(t2b + m * PITCH_B + n * 2) = (_Float16)t2;
        }
    };

    // GEMM2 epilogue: t1 = acc*(1-a1^2); fold W1 columns into grad partials.
    auto gemm2_epi = [&](v8f acc, int nt) {
        const int n = nt * 16 + l15;
        const float w10 = W1s[2 * n + 0], w11 = W1s[2 * n + 1];
#pragma unroll
        for (int d = 0; d < 8; ++d) {
            int m = d + 8 * half;
            float a1v = (float)(*(const _Float16*)(a1b + m * PITCH_B + n * 2));
            float t1  = acc[d] * (1.0f - a1v * a1v);
            g0a[d] += t1 * w10;
            g1a[d] += t1 * w11;
        }
    };

    const char* at1 = a1b + abyte0;
    const char* at2 = t2b + abyte0;

    // ---- Stage 2: GEMM1  a2 = tanh(a1 @ W2^T + b2); build t2 in LDS -------
    for (int nt2 = 0; nt2 < 16; ++nt2) {
        {   // even nt -> buffer 0
            const int nt = 2 * nt2;
            issue_panel(w2h, nt + 1, 1, ldsBBuf, tid);      // nt+1 <= 31
            asm volatile("s_wait_asynccnt 4" ::: "memory"); // panel nt landed
            __syncthreads();
            gemm1_epi(mma_sweep(at1, bp0), nt);
            __syncthreads();
        }
        {   // odd nt -> buffer 1
            const int nt = 2 * nt2 + 1;
            if (nt < 31) issue_panel(w2h, nt + 1, 0, ldsBBuf, tid);
            else         issue_panel(w2th, 0, 0, ldsBBuf, tid); // GEMM2 panel 0
            asm volatile("s_wait_asynccnt 4" ::: "memory");
            __syncthreads();
            gemm1_epi(mma_sweep(at1, bp1), nt);
            __syncthreads();
        }
    }
    asm volatile("s_wait_dscnt 0" ::: "memory");      // t2 tile visible

    // ---- Stage 3: GEMM2  t1 = (t2 @ W2) * (1 - a1^2); fold in W1 ----------
    for (int nt2 = 0; nt2 < 16; ++nt2) {
        {   // even nt -> buffer 0
            const int nt = 2 * nt2;
            issue_panel(w2th, nt + 1, 1, ldsBBuf, tid);     // nt+1 <= 31
            asm volatile("s_wait_asynccnt 4" ::: "memory");
            __syncthreads();
            gemm2_epi(mma_sweep(at2, bp0), nt);
            __syncthreads();
        }
        {   // odd nt -> buffer 1
            const int nt = 2 * nt2 + 1;
            if (nt < 31) {
                issue_panel(w2th, nt + 1, 0, ldsBBuf, tid);
                asm volatile("s_wait_asynccnt 4" ::: "memory");
            } else {
                asm volatile("s_wait_asynccnt 0" ::: "memory");
            }
            __syncthreads();
            gemm2_epi(mma_sweep(at2, bp1), nt);
            __syncthreads();
        }
    }

    // ---- Stage 4: reduce over the 16 lanes owning each row half -----------
#pragma unroll
    for (int d = 0; d < 8; ++d)
        for (int msk = 1; msk < 16; msk <<= 1) {
            vacc[d] += __shfl_xor(vacc[d], msk, 32);
            g0a[d]  += __shfl_xor(g0a[d],  msk, 32);
            g1a[d]  += __shfl_xor(g1a[d],  msk, 32);
        }
    float* redw = red + wave * 48;
    if (l15 == 0) {
#pragma unroll
        for (int d = 0; d < 8; ++d) {
            int m = d + 8 * half;
            redw[m * 3 + 0] = vacc[d];
            redw[m * 3 + 1] = g0a[d];
            redw[m * 3 + 2] = g1a[d];
        }
    }
    asm volatile("s_wait_dscnt 0" ::: "memory");

    // ---- Stage 5: dynamics + closed-form CLF-QP, one lane per row ---------
    if (lane < 16) {
        const int m = lane;
        const int g = tile * 16 + m;
        float V  = 0.5f * redw[m * 3 + 0];
        float g0 = redw[m * 3 + 1];
        float g1 = redw[m * 3 + 2];
        float x0 = x[2 * g], x1 = x[2 * g + 1];
        // f(x) with low-inertia model:  m*g*l/I = 9.81/1.05,  d/I = 0.1/1.05
        float sn = __builtin_amdgcn_sinf(x0 * 0.15915494309189535f); // x/(2pi)
        float f1 = 9.342857142857143f * sn - 0.09523809523809523f * x1;
        float LfV = g0 * x1 + g1 * f1;
        float LgV = g1 * 0.9523809523809523f;         // g = [0, 1/I]
        float unom = -(x0 * K[0] + x1 * K[1]);
        float s   = LfV + LgV * unom + V;             // lambda = 1
        float den = LgV * LgV + 0.05f;                // 1/(2p), p = 10
        float mu  = fmaxf(s, 0.0f) / den;
        float u   = unom - mu * LgV;
        float r   = mu * 0.05f;
        float Vd  = LfV + LgV * u;
        out[g]               = u;
        out[B_TOTAL + g]     = r;
        out[2 * B_TOTAL + g] = V;
        out[3 * B_TOTAL + g] = Vd;
    }
}

extern "C" void kernel_launch(void* const* d_in, const int* in_sizes, int n_in,
                              void* d_out, int out_size, void* d_ws, size_t ws_size,
                              hipStream_t stream) {
    (void)in_sizes; (void)n_in; (void)out_size; (void)ws_size;
    const float* x  = (const float*)d_in[0];
    const float* W1 = (const float*)d_in[1];
    const float* b1 = (const float*)d_in[2];
    const float* W2 = (const float*)d_in[3];
    const float* b2 = (const float*)d_in[4];
    const float* K  = (const float*)d_in[5];

    _Float16* w2h  = (_Float16*)d_ws;           // 512 KB
    _Float16* w2th = w2h + 512 * 512;           // 512 KB

    convert_w2<<<1024, 256, 0, stream>>>(W2, w2h, w2th);
    clf_qp_fused<<<NBLOCKS, WPB * 32, LDS_TOTAL, stream>>>(
        x, W1, b1, b2, K, w2h, w2th, (float*)d_out);
}
